// IXLinx8B_37366215475481
// MI455X (gfx1250) — matrix operand order repaired and checked by reading, not compiled, not run on hardware
//
#include <hip/hip_runtime.h>
#include <math.h>

// ---------------- types for WMMA ----------------
typedef __bf16 bf16_t;
typedef bf16_t v16bf __attribute__((ext_vector_type(16)));
typedef bf16_t v8bf  __attribute__((ext_vector_type(8)));
typedef bf16_t v2bf  __attribute__((ext_vector_type(2)));
typedef float  v8f   __attribute__((ext_vector_type(8)));

#define BETA_REG 0.05f

__device__ __forceinline__ float ereg_gelu_f(float xv) {
    float p   = 1.0f / (1.0f + __expf(-xv));
    float ent = -(p * __logf(p + 1e-9f) + (1.0f - p) * __logf(1.0f - p + 1e-9f));
    float g   = 0.5f * xv * (1.0f + erff(xv * 0.7071067811865475f));
    return g - BETA_REG * ent;
}

// =================================================================
// GEMM via v_wmma_f32_16x16x32_bf16,  C = act(A*Bw + bias) (+ resid)
// REQUIREMENT: K % 32 == 0, N % 64 == 0 (caller pads).
// Block: 128 threads (4 wave32). Block tile 64(M) x 64(N); each wave
// computes a 16x64 strip: FOUR accumulators sharing ONE A fragment
// (4 v_wmma per K-step, 4x A reuse). B K-slice staged TRANSPOSED
// (sBt[n][k], bf16) in LDS; staging packs two adjacent K rows into
// one bf16 pair -> single ds_store_b32 per pair; lane B fragments are
// contiguous 16B runs -> ds_load_b128. A fragment = two contiguous
// 8-float runs per lane -> global_load_b128 pairs.
// Output rows remapped: m -> base + (m/rpb)*bstride + (m%rpb)*ldc.
// =================================================================
__global__ __launch_bounds__(128)
void gemm_wmma_kernel(const float* __restrict__ A, int lda,
                      const float* __restrict__ Bw, int ldb,
                      const float* __restrict__ bias,
                      const float* __restrict__ resid,
                      float* __restrict__ C, int ldc,
                      int M, int N, int K, int act,
                      int rpb, long long bstride)
{
    __shared__ __align__(16) bf16_t sBt[64][32];   // [n][k], 4 KB

    const int tid   = threadIdx.x;
    const int lane  = tid & 31;
    const int wave  = tid >> 5;
    const int tileN = blockIdx.x * 64;
    const int tileM = blockIdx.y * 64 + wave * 16;

    const int n_lane = lane & 15;     // column in 16-wide subtile / A row
    const int half   = lane >> 4;     // K-half select per documented layout

    // clamped A row pointer (branch-free tail handling)
    int arow = tileM + n_lane;
    arow = (arow < M) ? arow : (M - 1);
    const float* __restrict__ aptr = A + (long long)arow * lda;

    // B staging: thread t loads K rows 2g,2g+1 and cols tileN+snn..+7
    const int g   = tid >> 3;         // 0..15  -> k rows 2g, 2g+1
    const int snn = (tid & 7) * 8;    // 0..56 step 8

    v8f acc[4] = {{}, {}, {}, {}};
    const int KT = K >> 5;

    for (int kt = 0; kt < KT; ++kt) {
        const int k0 = kt << 5;

        // ---- stage B (32k x 64n) fp32 -> bf16, transposed, into LDS ----
        const float* bp0 = Bw + (long long)(k0 + 2 * g) * ldb + (tileN + snn);
        const float* bp1 = bp0 + ldb;
        const float4 r0a = ((const float4*)bp0)[0];
        const float4 r0b = ((const float4*)bp0)[1];
        const float4 r1a = ((const float4*)bp1)[0];
        const float4 r1b = ((const float4*)bp1)[1];
        // pack [k=2g, k=2g+1] pairs -> one b32 store per column
        *(v2bf*)&sBt[snn + 0][2 * g] = (v2bf){(bf16_t)r0a.x, (bf16_t)r1a.x};
        *(v2bf*)&sBt[snn + 1][2 * g] = (v2bf){(bf16_t)r0a.y, (bf16_t)r1a.y};
        *(v2bf*)&sBt[snn + 2][2 * g] = (v2bf){(bf16_t)r0a.z, (bf16_t)r1a.z};
        *(v2bf*)&sBt[snn + 3][2 * g] = (v2bf){(bf16_t)r0a.w, (bf16_t)r1a.w};
        *(v2bf*)&sBt[snn + 4][2 * g] = (v2bf){(bf16_t)r0b.x, (bf16_t)r1b.x};
        *(v2bf*)&sBt[snn + 5][2 * g] = (v2bf){(bf16_t)r0b.y, (bf16_t)r1b.y};
        *(v2bf*)&sBt[snn + 6][2 * g] = (v2bf){(bf16_t)r0b.z, (bf16_t)r1b.z};
        *(v2bf*)&sBt[snn + 7][2 * g] = (v2bf){(bf16_t)r0b.w, (bf16_t)r1b.w};
        if (k0 + 32 < K)
            __builtin_prefetch(bp0 + (long long)32 * ldb, 0, 0);
        __syncthreads();

        // ---- A fragment: lanes 0-15 K={0..7,16..23}, lanes 16-31 K={8..15,24..31}
        const float4* ap1 = (const float4*)(aptr + k0 + half * 8);
        const float4* ap2 = (const float4*)(aptr + k0 + 16 + half * 8);
        const float4 a0 = ap1[0], a1 = ap1[1];
        const float4 a2 = ap2[0], a3 = ap2[1];
        v16bf af;
        af[0]  = (bf16_t)a0.x; af[1]  = (bf16_t)a0.y;
        af[2]  = (bf16_t)a0.z; af[3]  = (bf16_t)a0.w;
        af[4]  = (bf16_t)a1.x; af[5]  = (bf16_t)a1.y;
        af[6]  = (bf16_t)a1.z; af[7]  = (bf16_t)a1.w;
        af[8]  = (bf16_t)a2.x; af[9]  = (bf16_t)a2.y;
        af[10] = (bf16_t)a2.z; af[11] = (bf16_t)a2.w;
        af[12] = (bf16_t)a3.x; af[13] = (bf16_t)a3.y;
        af[14] = (bf16_t)a3.z; af[15] = (bf16_t)a3.w;

        // ---- 4 B fragments from LDS, 4 WMMAs reusing the A fragment ----
        const int kb = half * 8;
        #pragma unroll
        for (int c = 0; c < 4; ++c) {
            const int ncol = c * 16 + n_lane;
            const v8bf p0 = *(const v8bf*)&sBt[ncol][kb];
            const v8bf p1 = *(const v8bf*)&sBt[ncol][kb + 16];
            v16bf bfr;
            #pragma unroll
            for (int i = 0; i < 8; ++i) { bfr[i] = p0[i]; bfr[i + 8] = p1[i]; }
            acc[c] = __builtin_amdgcn_wmma_f32_16x16x32_bf16(
                         false, af, false, bfr, (short)0, acc[c], false, false);
        }
        __syncthreads();
    }

    // ---- epilogue: VGPR r -> M = r + 8*(lane>=16), N = lane&15 ----
    #pragma unroll
    for (int r = 0; r < 8; ++r) {
        const int mrow = tileM + r + half * 8;
        if (mrow < M) {
            const int bidx = mrow / rpb;
            const int rrow = mrow - bidx * rpb;
            const long long rowoff = (long long)bidx * bstride +
                                     (long long)rrow * ldc;
            #pragma unroll
            for (int c = 0; c < 4; ++c) {
                const int ncol = tileN + c * 16 + n_lane;
                float v = acc[c][r];
                if (bias) v += bias[ncol];
                if (act == 1) v = ereg_gelu_f(v);
                if (resid) v += resid[rowoff + ncol];
                C[rowoff + ncol] = v;
            }
        }
    }
}

// ================= token embedding gather =================
__global__ void tok_kernel(const int* __restrict__ text,
                           const float* __restrict__ emb,
                           float* __restrict__ h,
                           int B, int T, int S, int D)
{
    long long idx = (long long)blockIdx.x * blockDim.x + threadIdx.x;
    const long long total = (long long)B * T * D;
    if (idx >= total) return;
    const int d = (int)(idx % D);
    const long long bt = idx / D;
    const int t = (int)(bt % T);
    const int b = (int)(bt / T);
    const int tok = text[b * T + t];
    h[((long long)b * S + t) * D + d] = emb[(long long)tok * D + d];
}

// ================= image patch extraction (B,196,768) =================
__global__ void patch_kernel(const float* __restrict__ img,
                             float* __restrict__ patches, int B)
{
    long long idx = (long long)blockIdx.x * blockDim.x + threadIdx.x;
    const long long total = (long long)B * 196 * 768;
    if (idx >= total) return;
    const int k = (int)(idx % 768);
    const long long bp = idx / 768;
    const int p = (int)(bp % 196);
    const int b = (int)(bp / 196);
    const int c = k >> 8, rem = k & 255, dy = rem >> 4, dx = rem & 15;
    const int ph = p / 14, pw = p % 14;
    patches[idx] = img[(((long long)b * 3 + c) * 224 + (ph * 16 + dy)) * 224 +
                       (pw * 16 + dx)];
}

// ========= audio frame extraction, K padded 400 -> 416 (B,79,416) =========
__global__ void frame_kernel(const float* __restrict__ audio,
                             float* __restrict__ frames, int B)
{
    long long idx = (long long)blockIdx.x * blockDim.x + threadIdx.x;
    const long long total = (long long)B * 79 * 416;
    if (idx >= total) return;
    const int k = (int)(idx % 416);
    const long long bf = idx / 416;
    const int f = (int)(bf % 79);
    const int b = (int)(bf / 79);
    frames[idx] = (k < 400) ? audio[(long long)b * 16000 + f * 200 + k] : 0.0f;
}

// ========= zero-padded copy of aud_w: (400,768) -> (416,768) =========
__global__ void audpad_kernel(const float* __restrict__ aw,
                              float* __restrict__ awp)
{
    const int idx = blockIdx.x * blockDim.x + threadIdx.x;
    if (idx >= 416 * 768) return;
    const int r = idx / 768;
    awp[idx] = (r < 400) ? aw[idx] : 0.0f;
}

// ================= LayerNorm (one block per row) =================
__global__ __launch_bounds__(256)
void ln_kernel(const float* __restrict__ in,
               const float* __restrict__ g, const float* __restrict__ bb,
               float* __restrict__ out, int rows, int D)
{
    const int row = blockIdx.x;
    if (row >= rows) return;
    const float* p = in + (long long)row * D;
    float s = 0.f, s2 = 0.f;
    for (int i = threadIdx.x; i < D; i += blockDim.x) {
        const float v = p[i]; s += v; s2 += v * v;
    }
    for (int off = 16; off > 0; off >>= 1) {
        s  += __shfl_down(s,  off, 32);
        s2 += __shfl_down(s2, off, 32);
    }
    __shared__ float rs[8], rs2[8];
    const int wid = threadIdx.x >> 5, lane = threadIdx.x & 31;
    if (lane == 0) { rs[wid] = s; rs2[wid] = s2; }
    __syncthreads();
    if (threadIdx.x == 0) {
        float a = 0.f, b2 = 0.f;
        const int nw = blockDim.x >> 5;
        for (int i = 0; i < nw; ++i) { a += rs[i]; b2 += rs2[i]; }
        rs[0] = a; rs2[0] = b2;
    }
    __syncthreads();
    const float mean = rs[0] / D;
    const float var  = rs2[0] / D - mean * mean;
    const float inv  = rsqrtf(var + 1e-5f);
    float* o = out + (long long)row * D;
    for (int i = threadIdx.x; i < D; i += blockDim.x)
        o[i] = (p[i] - mean) * inv * g[i] + bb[i];
}

// ====== final LN that also packs only rows s<T into (B*T, D) ======
__global__ __launch_bounds__(256)
void lnf_pack_kernel(const float* __restrict__ h,
                     const float* __restrict__ g, const float* __restrict__ bb,
                     float* __restrict__ out, int B, int T, int S, int D)
{
    const int row = blockIdx.x;                 // 0..B*T-1
    if (row >= B * T) return;
    const int b = row / T, t = row % T;
    const float* p = h + ((long long)b * S + t) * D;
    float s = 0.f, s2 = 0.f;
    for (int i = threadIdx.x; i < D; i += blockDim.x) {
        const float v = p[i]; s += v; s2 += v * v;
    }
    for (int off = 16; off > 0; off >>= 1) {
        s  += __shfl_down(s,  off, 32);
        s2 += __shfl_down(s2, off, 32);
    }
    __shared__ float rs[8], rs2[8];
    const int wid = threadIdx.x >> 5, lane = threadIdx.x & 31;
    if (lane == 0) { rs[wid] = s; rs2[wid] = s2; }
    __syncthreads();
    if (threadIdx.x == 0) {
        float a = 0.f, b2 = 0.f;
        for (int i = 0; i < 8; ++i) { a += rs[i]; b2 += rs2[i]; }
        rs[0] = a; rs2[0] = b2;
    }
    __syncthreads();
    const float mean = rs[0] / D;
    const float var  = rs2[0] / D - mean * mean;
    const float inv  = rsqrtf(var + 1e-5f);
    float* o = out + (long long)row * D;
    for (int i = threadIdx.x; i < D; i += blockDim.x)
        o[i] = (p[i] - mean) * inv * g[i] + bb[i];
}

// ================= sequential SSM scan =================
__global__ void scan_kernel(const float* __restrict__ inp,
                            const float* __restrict__ gm,
                            const float* __restrict__ bt,
                            float* __restrict__ states,
                            int B, int S, int H)
{
    const int idx = blockIdx.x * blockDim.x + threadIdx.x;  // b*H + h
    if (idx >= B * H) return;
    const int b = idx / H, hh = idx % H;
    const float g = gm[hh], be = bt[hh];
    const float* ip = inp    + (long long)b * S * H + hh;
    float*       sp = states + (long long)b * S * H + hh;
    float s = 0.f;
    for (int t = 0; t < S; ++t) {
        s = tanhf(ip[(long long)t * H] + s * g + be);
        sp[(long long)t * H] = s;
    }
}

// ================= host orchestration =================
static void launch_gemm(const float* A, int lda, const float* Bw, int ldb,
                        const float* bias, const float* resid,
                        float* C, int ldc, int M, int N, int K, int act,
                        int rpb, long long bstride, hipStream_t st)
{
    dim3 grid(N / 64, (M + 63) / 64);
    gemm_wmma_kernel<<<grid, 128, 0, st>>>(A, lda, Bw, ldb, bias, resid,
                                           C, ldc, M, N, K, act, rpb, bstride);
}

extern "C" void kernel_launch(void* const* d_in, const int* in_sizes, int n_in,
                              void* d_out, int out_size, void* d_ws, size_t ws_size,
                              hipStream_t stream)
{
    (void)in_sizes; (void)n_in; (void)out_size; (void)ws_size;

    const int B = 4, T = 512, V = 32000, D = 768, H = 1536, R = 192, Iw = 1536, L = 12;
    const int NP = 196, NF = 79, AWp = 416;   // audio K padded to 416 (13*32)
    const int S = T + NP + NF;                // 787
    const int M = B * S;                      // 3148

    const int*   text       = (const int*)  d_in[0];
    const float* image      = (const float*)d_in[1];
    const float* audio      = (const float*)d_in[2];
    const float* text_embed = (const float*)d_in[3];
    const float* img_w      = (const float*)d_in[4];
    const float* img_b      = (const float*)d_in[5];
    const float* aud_w      = (const float*)d_in[6];
    const float* aud_b      = (const float*)d_in[7];
    const float* ln1_g      = (const float*)d_in[8];
    const float* ln1_b      = (const float*)d_in[9];
    const float* ln2_g      = (const float*)d_in[10];
    const float* ln2_b      = (const float*)d_in[11];
    const float* in_w       = (const float*)d_in[12];
    const float* in_b       = (const float*)d_in[13];
    const float* out_w      = (const float*)d_in[14];
    const float* out_b      = (const float*)d_in[15];
    const float* ssm_gamma  = (const float*)d_in[16];
    const float* ssm_beta   = (const float*)d_in[17];
    const float* fc1_u      = (const float*)d_in[18];
    const float* fc1_v      = (const float*)d_in[19];
    const float* fc1_b      = (const float*)d_in[20];
    const float* fc2_u      = (const float*)d_in[21];
    const float* fc2_v      = (const float*)d_in[22];
    const float* fc2_b      = (const float*)d_in[23];
    const float* lnf_g      = (const float*)d_in[24];
    const float* lnf_b      = (const float*)d_in[25];
    const float* head_w     = (const float*)d_in[26];
    const float* head_b     = (const float*)d_in[27];
    float* out = (float*)d_out;

    // ---- workspace layout (floats) ----
    float* ws = (float*)d_ws;
    float* h        = ws; ws += (long long)M * D;
    float* x        = ws; ws += (long long)M * D;
    float* inp      = ws; ws += (long long)M * H;
    float* states   = ws; ws += (long long)M * H;
    float* z        = ws; ws += (long long)M * Iw;
    float* t1       = ws; ws += (long long)M * R;
    float* xf       = ws; ws += (long long)B * T * D;
    float* patches  = ws; ws += (long long)B * NP * 768;
    float* frames   = ws; ws += (long long)B * NF * AWp;
    float* aud_wpad = ws; ws += (long long)AWp * D;

    const long long hstride = (long long)S * D;

    // ---- build h = concat(tok, img_feat, aud_feat) along sequence ----
    {
        long long tot = (long long)B * T * D;
        tok_kernel<<<(unsigned)((tot + 255) / 256), 256, 0, stream>>>(
            text, text_embed, h, B, T, S, D);
    }
    {
        long long tot = (long long)B * NP * 768;
        patch_kernel<<<(unsigned)((tot + 255) / 256), 256, 0, stream>>>(
            image, patches, B);
        launch_gemm(patches, 768, img_w, D, img_b, nullptr,
                    h + (long long)T * D, D, B * NP, D, 768, 0, NP, hstride, stream);
    }
    {
        audpad_kernel<<<(416 * 768 + 255) / 256, 256, 0, stream>>>(aud_w, aud_wpad);
        long long tot = (long long)B * NF * AWp;
        frame_kernel<<<(unsigned)((tot + 255) / 256), 256, 0, stream>>>(
            audio, frames, B);
        launch_gemm(frames, AWp, aud_wpad, D, aud_b, nullptr,
                    h + (long long)(T + NP) * D, D, B * NF, D, AWp, 0, NF, hstride, stream);
    }

    // ---- transformer blocks ----
    for (int l = 0; l < L; ++l) {
        // SSM branch
        ln_kernel<<<M, 256, 0, stream>>>(h, ln1_g + l * D, ln1_b + l * D, x, M, D);
        launch_gemm(x, D, in_w + (long long)l * D * H, H, in_b + l * H, nullptr,
                    inp, H, M, H, D, 0, M, 0, stream);
        scan_kernel<<<(B * H + 255) / 256, 256, 0, stream>>>(
            inp, ssm_gamma + l * H, ssm_beta + l * H, states, B, S, H);
        launch_gemm(states, H, out_w + (long long)l * H * D, D, out_b + l * D, h,
                    h, D, M, D, H, 0, M, 0, stream);

        // low-rank FFN branch with entropy-regularized GELU
        ln_kernel<<<M, 256, 0, stream>>>(h, ln2_g + l * D, ln2_b + l * D, x, M, D);
        launch_gemm(x, D, fc1_u + (long long)l * D * R, R, nullptr, nullptr,
                    t1, R, M, R, D, 0, M, 0, stream);
        launch_gemm(t1, R, fc1_v + (long long)l * R * Iw, Iw, fc1_b + l * Iw, nullptr,
                    z, Iw, M, Iw, R, /*act=*/1, M, 0, stream);
        launch_gemm(z, Iw, fc2_u + (long long)l * Iw * R, R, nullptr, nullptr,
                    t1, R, M, R, Iw, 0, M, 0, stream);
        launch_gemm(t1, R, fc2_v + (long long)l * R * D, D, fc2_b + l * D, h,
                    h, D, M, D, R, 0, M, 0, stream);
    }

    // ---- final LN (only the T text positions) + vocab head ----
    lnf_pack_kernel<<<B * T, 256, 0, stream>>>(h, lnf_g, lnf_b, xf, B, T, S, D);
    launch_gemm(xf, D, head_w, V, head_b, nullptr,
                out, V, B * T, V, D, 0, B * T, 0, stream);
}
